// GenBlock_90099823935853
// MI455X (gfx1250) — matrix operand order repaired and checked by reading, not compiled, not run on hardware
//
#include <hip/hip_runtime.h>
#include <hip/hip_bf16.h>

// ---------------------------------------------------------------------------
// GenBlock forward for MI455X (gfx1250), wave32 + v_wmma_f32_16x16x32_f16.
// All channel contractions (upsample convT, two AdaMod convs) run as WMMA
// GEMMs with f16 operands / f32 accumulation; modulation folded into
// A-scaling, demodulation folded into the epilogue.  LDS staging uses
// GLOBAL_LOAD_ASYNC_TO_LDS_B128 (ASYNCcnt) with a double-buffered pipeline
// when the toolchain exposes the async builtins.
// ---------------------------------------------------------------------------

typedef _Float16 v4h  __attribute__((ext_vector_type(4)));
typedef _Float16 v8h  __attribute__((ext_vector_type(8)));
typedef _Float16 v16h __attribute__((ext_vector_type(16)));
typedef float    v8f  __attribute__((ext_vector_type(8)));
typedef int      v4i  __attribute__((ext_vector_type(4)));

static constexpr int C  = 256;   // channels (Cin == Cout)
static constexpr int L  = 256;   // latent
static constexpr int NB = 16;    // batch
static constexpr int W1 = 2048;  // input width
static constexpr int W2 = 4096;  // output width
static constexpr int A_STR = 264;  // LDS A row stride (halves), 16B aligned
static constexpr int B_STR = 40;   // LDS B row stride (halves), 16B aligned
static constexpr float ALPHA_LR = 0.2f;
static constexpr float INV_RT2  = 0.70710678118654752440f;

#if defined(__gfx1250__) && __has_builtin(__builtin_amdgcn_global_load_async_to_lds_b128)
#define HAS_ASYNC_LDS 1
#else
#define HAS_ASYNC_LDS 0
#endif

typedef __attribute__((address_space(1))) v4i* gv4i_p;  // global (device) int4*
typedef __attribute__((address_space(3))) v4i* lv4i_p;  // LDS int4*

__device__ __forceinline__ void async_wait_all() {
#if HAS_ASYNC_LDS
#if __has_builtin(__builtin_amdgcn_s_wait_asynccnt)
    __builtin_amdgcn_s_wait_asynccnt(0);
#else
    asm volatile("s_wait_asynccnt 0x0" ::: "memory");
#endif
#endif
}

// 16-byte global -> LDS copy: async DMA when available, else through VGPRs.
__device__ __forceinline__ void cp_b128(const _Float16* g, _Float16* l) {
#if HAS_ASYNC_LDS
    void* gp = (void*)g;   // strip const first; then retype + addrspace
    void* lp = (void*)l;
    __builtin_amdgcn_global_load_async_to_lds_b128((gv4i_p)gp, (lv4i_p)lp, 0, 0);
#else
    *(v8h*)l = *(const v8h*)g;
#endif
}

__device__ __forceinline__ float softplusf(float x) {
    // log1p(exp(x)) stable form: max(x,0) + log1p(exp(-|x|))
    return fmaxf(x, 0.f) + log1pf(expf(-fabsf(x)));
}
__device__ __forceinline__ float lreluf(float x) {
    return x > 0.f ? x : ALPHA_LR * x;
}

// ---------------------------------------------------------------------------
// s1 = softplus(ltnt@s1w + s1b) + 1 ; s2 likewise ; ns = softplus(...) ;
// nb = ltnt@nb_w + nb_b.   One block per sample, one thread per channel.
// ---------------------------------------------------------------------------
__global__ __launch_bounds__(256)
void scales_kernel(const float* __restrict__ ltnt,
                   const float* __restrict__ s1w, const float* __restrict__ s1b,
                   const float* __restrict__ s2w, const float* __restrict__ s2b,
                   const float* __restrict__ nbw, const float* __restrict__ nbb,
                   const float* __restrict__ nsw, const float* __restrict__ nsb,
                   float* __restrict__ s1, float* __restrict__ s2,
                   float* __restrict__ nb, float* __restrict__ ns) {
    __shared__ float lt[L];
    const int n = blockIdx.x, c = threadIdx.x;
    lt[c] = ltnt[n * L + c];
    __syncthreads();
    float a1 = 0.f, a2 = 0.f, ab = 0.f, as = 0.f;
    for (int l = 0; l < L; ++l) {
        const float v = lt[l];
        a1 = fmaf(v, s1w[l * C + c], a1);
        a2 = fmaf(v, s2w[l * C + c], a2);
        ab = fmaf(v, nbw[l * C + c], ab);
        as = fmaf(v, nsw[l * C + c], as);
    }
    s1[n * C + c] = softplusf(a1 + s1b[c]) + 1.f;
    s2[n * C + c] = softplusf(a2 + s2b[c]) + 1.f;
    nb[n * C + c] = ab + nbb[c];
    ns[n * C + c] = softplusf(as + nsb[c]);
}

// ---------------------------------------------------------------------------
// d[n,co] = 1/sqrt( sum_{k,ci} (k[k,ci,co]*s[n,ci])^2 + 1e-8 )
// ---------------------------------------------------------------------------
__global__ __launch_bounds__(256)
void demod_kernel(const float* __restrict__ k1, const float* __restrict__ k2,
                  const float* __restrict__ s1, const float* __restrict__ s2,
                  float* __restrict__ d1, float* __restrict__ d2) {
    __shared__ float sA[C];
    __shared__ float sB[C];
    const int n = blockIdx.x, c = threadIdx.x;
    sA[c] = s1[n * C + c];
    sB[c] = s2[n * C + c];
    __syncthreads();
    float acc1 = 0.f, acc2 = 0.f;
    for (int ci = 0; ci < C; ++ci) {
        float q1 = 0.f, q2 = 0.f;
        #pragma unroll
        for (int k = 0; k < 3; ++k) {
            const float a = k1[(k * C + ci) * C + c];
            const float b = k2[(k * C + ci) * C + c];
            q1 = fmaf(a, a, q1);
            q2 = fmaf(b, b, q2);
        }
        acc1 = fmaf(sA[ci] * sA[ci], q1, acc1);
        acc2 = fmaf(sB[ci] * sB[ci], q2, acc2);
    }
    d1[n * C + c] = 1.f / sqrtf(acc1 + 1e-8f);
    d2[n * C + c] = 1.f / sqrtf(acc2 + 1e-8f);
}

// ---------------------------------------------------------------------------
// Weight convert to f16, transposed so that BT[k][co][ci] is the B matrix
// with contiguous ci (WMMA K) per co row.
//   upT[k][co][ci] = up_k[k][co][ci]      (TF (k,out,in) already matches)
//   k1T[k][co][ci] = k1[k][ci][co]
// ---------------------------------------------------------------------------
__global__ __launch_bounds__(256)
void wconv_kernel(const float* __restrict__ upk, const float* __restrict__ k1,
                  const float* __restrict__ k2, _Float16* __restrict__ upT,
                  _Float16* __restrict__ k1T, _Float16* __restrict__ k2T) {
    const int idx = blockIdx.x * 256 + threadIdx.x;      // < 3 * 3*65536
    const int which = idx / (3 * C * C);
    const int rem = idx % (3 * C * C);
    const int k = rem / (C * C);
    const int rc = rem % (C * C);
    const int co = rc >> 8;
    const int ci = rc & 255;
    if (which == 0)      upT[rem] = (_Float16)upk[(k * C + co) * C + ci];
    else if (which == 1) k1T[rem] = (_Float16)k1[(k * C + ci) * C + co];
    else                 k2T[rem] = (_Float16)k2[(k * C + ci) * C + co];
}

__global__ __launch_bounds__(256)
void dconv_kernel(const float* __restrict__ src, _Float16* __restrict__ dst, int n4) {
    const int i = blockIdx.x * 256 + threadIdx.x;
    if (i >= n4) return;
    const float4 v = ((const float4*)src)[i];
    v4h h;
    h[0] = (_Float16)v.x; h[1] = (_Float16)v.y;
    h[2] = (_Float16)v.z; h[3] = (_Float16)v.w;
    ((v4h*)dst)[i] = h;
}

// ---------------------------------------------------------------------------
// Tiled WMMA conv core. Block = 256 threads (8 waves), tile = 64 rows x 256
// couts for sample n = blockIdx.y.  Waves: wm in {0,1} (32 rows), wn in
// {0..3} (64 couts).  Each wave: 2 M-tiles x 4 N-tiles of 16x16 -> 8 accs.
// K loop: taps (shifted A rows from a 66-row halo in LDS) x 8 chunks of 32 ci,
// with the B chunk double-buffered: async prefetch of chunk kc+1 overlaps the
// 8 WMMAs of chunk kc; one barrier per chunk.
//
// MODE 0 (upsample convT, grid.z = 2 phases):
//   phase 0 (even t): taps {w0 @ +0, w2 @ -1};  phase 1 (odd t): {w1 @ +0}
//   out: x0s_h = f16( acc * s1[n,co] )               at t = 2*(j)+phase
// MODE 1 (modconv1): y = acc*d1 ; lrelu ; out f16( y * s2[n,co] )
// MODE 2 (modconv2): y = acc*d2 + nb + nois*(ns+1e-8) ; lrelu ;
//   out f32 -> d_out (x) and f16 copy for toImage.
// ---------------------------------------------------------------------------
template <int MODE>
__global__ __launch_bounds__(256)
void conv_wmma_kernel(const _Float16* __restrict__ Ah, int Arows,
                      const _Float16* __restrict__ BT,
                      const float* __restrict__ e0, const float* __restrict__ e1,
                      const float* __restrict__ e2, const float* __restrict__ nois,
                      _Float16* __restrict__ OutH, float* __restrict__ OutF) {
    __shared__ __align__(16) _Float16 As[66 * A_STR];
    __shared__ __align__(16) _Float16 Bs[2 * C * B_STR];   // double buffer

    const int tid = threadIdx.x;
    const int n = blockIdx.y;
    const int phase = blockIdx.z;
    const int t0 = blockIdx.x * 64;   // output row base (j base for MODE 0)

    // ---- halo: rows t0-1 .. t0+64 (66 rows) of f16 A -> LDS (async) ----
    {
        const int sub = tid & 31;      // 16B column chunk
        const int rg = tid >> 5;       // starting row (uniform per wave)
        for (int rr = rg; rr < 66; rr += 8) {
            const int g = t0 - 1 + rr;
            _Float16* dst = As + rr * A_STR + sub * 8;
            if (g >= 0 && g < Arows) {
                cp_b128(Ah + ((size_t)n * Arows + g) * C + sub * 8, dst);
            } else {
                v8h z;
                #pragma unroll
                for (int j = 0; j < 8; ++j) z[j] = (_Float16)0.f;
                *(v8h*)dst = z;
            }
        }
    }

    const int lane = tid & 31, wave = tid >> 5;
    const int wm = wave & 1, wn = wave >> 1;
    const int h = lane >> 4, l16 = lane & 15;

    v8f acc[2][4];
    #pragma unroll
    for (int mi = 0; mi < 2; ++mi)
        #pragma unroll
        for (int ni = 0; ni < 4; ++ni)
            #pragma unroll
            for (int r = 0; r < 8; ++r) acc[mi][ni][r] = 0.f;

    int ntap = 3;
    if (MODE == 0) ntap = (phase == 0) ? 2 : 1;

    for (int tp = 0; tp < ntap; ++tp) {
        int wt, lro;   // weight tap index, LDS row offset (src_off + 1)
        if (MODE == 0) {
            if (phase == 0) { wt = (tp == 0) ? 0 : 2; lro = (tp == 0) ? 1 : 0; }
            else            { wt = 1; lro = 1; }
        } else { wt = tp; lro = tp; }
        const _Float16* BTt = BT + (size_t)wt * C * C;

        // previous readers of buffer 0 are done after this barrier
        __syncthreads();
        {   // issue chunk 0 -> buffer 0 (row co = tid, 32 ci = 64B)
            const _Float16* src = BTt + tid * C;
            _Float16* dst = Bs + tid * B_STR;
            #pragma unroll
            for (int j = 0; j < 4; ++j) cp_b128(src + j * 8, dst + j * 8);
        }

        #pragma unroll
        for (int kc = 0; kc < 8; ++kc) {      // 8 chunks of 32 ci
            async_wait_all();                 // current buffer's DMA complete
            __syncthreads();                  // all waves ready; prev reads done
            if (kc < 7) {                     // prefetch next chunk, overlaps WMMA
                const _Float16* src = BTt + tid * C + (kc + 1) * 32;
                _Float16* dst = Bs + ((kc + 1) & 1) * (C * B_STR) + tid * B_STR;
                #pragma unroll
                for (int j = 0; j < 4; ++j) cp_b128(src + j * 8, dst + j * 8);
            }
            const _Float16* Bcur = Bs + (kc & 1) * (C * B_STR);

            // B fragments: 32x16, lane(l16)=col; b[0..7]=K(8h..), b[8..15]=K(16+8h..)
            v16h bfr[4];
            #pragma unroll
            for (int ni = 0; ni < 4; ++ni) {
                const _Float16* bp = Bcur + (wn * 64 + ni * 16 + l16) * B_STR + h * 8;
                const v8h blo = *(const v8h*)(bp);
                const v8h bhi = *(const v8h*)(bp + 16);
                bfr[ni] = __builtin_shufflevector(blo, bhi, 0, 1, 2, 3, 4, 5, 6, 7,
                                                  8, 9, 10, 11, 12, 13, 14, 15);
            }
            // A fragments: 16x32, lane(l16)=row; a[0..7]=K(8h..), a[8..15]=K(16+8h..)
            #pragma unroll
            for (int mi = 0; mi < 2; ++mi) {
                const _Float16* ap =
                    As + (wm * 32 + mi * 16 + l16 + lro) * A_STR + kc * 32 + h * 8;
                const v8h alo = *(const v8h*)(ap);
                const v8h ahi = *(const v8h*)(ap + 16);
                const v16h afr = __builtin_shufflevector(alo, ahi, 0, 1, 2, 3, 4, 5, 6, 7,
                                                         8, 9, 10, 11, 12, 13, 14, 15);
                #pragma unroll
                for (int ni = 0; ni < 4; ++ni)
                    acc[mi][ni] = __builtin_amdgcn_wmma_f32_16x16x32_f16(
                        false, afr, false, bfr[ni], (short)0, acc[mi][ni], false, false);
            }
        }
    }

    // ---- epilogue: D tile element r -> row (r + 8h), col l16 ----
    #pragma unroll
    for (int mi = 0; mi < 2; ++mi) {
        #pragma unroll
        for (int ni = 0; ni < 4; ++ni) {
            const int co = wn * 64 + ni * 16 + l16;
            if (MODE == 0) {
                const float sv = e0[n * C + co];   // s1 (next layer's modulation)
                #pragma unroll
                for (int r = 0; r < 8; ++r) {
                    const int m = wm * 32 + mi * 16 + r + h * 8;
                    const int t = 2 * (t0 + m) + phase;
                    OutH[((size_t)n * W2 + t) * C + co] = (_Float16)(acc[mi][ni][r] * sv);
                }
            } else if (MODE == 1) {
                const float dv = e0[n * C + co];   // d1
                const float sv = e1[n * C + co];   // s2
                #pragma unroll
                for (int r = 0; r < 8; ++r) {
                    const int m = wm * 32 + mi * 16 + r + h * 8;
                    const int t = t0 + m;
                    float y = lreluf(acc[mi][ni][r] * dv);
                    OutH[((size_t)n * W2 + t) * C + co] = (_Float16)(y * sv);
                }
            } else {
                const float dv = e0[n * C + co];          // d2
                const float bv = e1[n * C + co];          // noise bias
                const float nsv = e2[n * C + co] + 1e-8f; // noise scale
                #pragma unroll
                for (int r = 0; r < 8; ++r) {
                    const int m = wm * 32 + mi * 16 + r + h * 8;
                    const int t = t0 + m;
                    float y = acc[mi][ni][r] * dv + bv +
                              nois[(size_t)n * W2 + t] * nsv;
                    y = lreluf(y);
                    const size_t oi = ((size_t)n * W2 + t) * C + co;
                    OutF[oi] = y;
                    OutH[oi] = (_Float16)y;
                }
            }
        }
    }
}

// ---------------------------------------------------------------------------
// im = (convT(img, imgup_k) + conv1d(x, img_k, SAME)) * 1/sqrt(2)
// ---------------------------------------------------------------------------
__global__ __launch_bounds__(256)
void to_image_kernel(const _Float16* __restrict__ xh, const float* __restrict__ img,
                     const float* __restrict__ img_k, const float* __restrict__ imgup_k,
                     float* __restrict__ outIm) {
    __shared__ float wk[3 * C];
    const int tid = threadIdx.x;
    for (int i = tid; i < 3 * C; i += 256) wk[i] = img_k[i];
    __syncthreads();
    const int n = blockIdx.y;
    const int t = blockIdx.x * 256 + tid;
    float acc = 0.f;
    #pragma unroll
    for (int k = 0; k < 3; ++k) {
        const int g = t + k - 1;
        if (g < 0 || g >= W2) continue;
        const v8h* row = (const v8h*)(xh + ((size_t)n * W2 + g) * C);
        const float* wr = wk + k * C;
        for (int c8 = 0; c8 < C / 8; ++c8) {
            const v8h v = row[c8];
            #pragma unroll
            for (int j = 0; j < 8; ++j)
                acc = fmaf((float)v[j], wr[c8 * 8 + j], acc);
        }
    }
    const float iu0 = imgup_k[0], iu1 = imgup_k[1], iu2 = imgup_k[2];
    float upv;
    if ((t & 1) == 0) {
        const int i = t >> 1;
        upv = img[n * W1 + i] * iu0 + ((i - 1) >= 0 ? img[n * W1 + i - 1] * iu2 : 0.f);
    } else {
        const int i = (t - 1) >> 1;
        upv = img[n * W1 + i] * iu1;
    }
    outIm[(size_t)n * W2 + t] = (acc + upv) * INV_RT2;
}

// ---------------------------------------------------------------------------
extern "C" void kernel_launch(void* const* d_in, const int* in_sizes, int n_in,
                              void* d_out, int out_size, void* d_ws, size_t ws_size,
                              hipStream_t stream) {
    (void)in_sizes; (void)n_in; (void)out_size; (void)ws_size;
    const float* data  = (const float*)d_in[0];
    const float* ltnt  = (const float*)d_in[1];
    const float* nois  = (const float*)d_in[2];
    const float* img   = (const float*)d_in[3];
    const float* up_k  = (const float*)d_in[4];
    const float* k1    = (const float*)d_in[5];
    const float* s1w   = (const float*)d_in[6];
    const float* s1b   = (const float*)d_in[7];
    const float* k2    = (const float*)d_in[8];
    const float* s2w   = (const float*)d_in[9];
    const float* s2b   = (const float*)d_in[10];
    const float* nbw   = (const float*)d_in[11];
    const float* nbb   = (const float*)d_in[12];
    const float* nsw   = (const float*)d_in[13];
    const float* nsb   = (const float*)d_in[14];
    const float* iupk  = (const float*)d_in[15];
    const float* imgk  = (const float*)d_in[16];

    char* ws = (char*)d_ws;
    float* s1 = (float*)(ws + 0);
    float* s2 = (float*)(ws + 16384);
    float* nb = (float*)(ws + 32768);
    float* ns = (float*)(ws + 49152);
    float* d1 = (float*)(ws + 65536);
    float* d2 = (float*)(ws + 81920);
    _Float16* upT   = (_Float16*)(ws + 98304);
    _Float16* k1T   = (_Float16*)(ws + 491520);
    _Float16* k2T   = (_Float16*)(ws + 884736);
    _Float16* dataH = (_Float16*)(ws + 1277952);              // 16 MiB
    _Float16* x0s   = (_Float16*)(ws + 18055168);             // 32 MiB
    _Float16* x1s   = (_Float16*)(ws + 51609600);             // 32 MiB
    _Float16* xh    = x0s;  // reuse: x0s dead after conv1

    float* outX  = (float*)d_out;
    float* outIm = outX + (size_t)NB * W2 * C;

    scales_kernel<<<NB, 256, 0, stream>>>(ltnt, s1w, s1b, s2w, s2b,
                                          nbw, nbb, nsw, nsb, s1, s2, nb, ns);
    demod_kernel<<<NB, 256, 0, stream>>>(k1, k2, s1, s2, d1, d2);
    wconv_kernel<<<(3 * 3 * C * C) / 256, 256, 0, stream>>>(up_k, k1, k2, upT, k1T, k2T);
    dconv_kernel<<<(NB * W1 * C / 4 + 255) / 256, 256, 0, stream>>>(data, dataH,
                                                                    NB * W1 * C / 4);
    // upsample convT: even/odd phases, 2048 j-rows each
    conv_wmma_kernel<0><<<dim3(W1 / 64, NB, 2), 256, 0, stream>>>(
        dataH, W1, upT, s1, nullptr, nullptr, nullptr, x0s, nullptr);
    // modconv1
    conv_wmma_kernel<1><<<dim3(W2 / 64, NB, 1), 256, 0, stream>>>(
        x0s, W2, k1T, d1, s2, nullptr, nullptr, x1s, nullptr);
    // modconv2 + noise + lrelu -> d_out (f32) + f16 copy
    conv_wmma_kernel<2><<<dim3(W2 / 64, NB, 1), 256, 0, stream>>>(
        x1s, W2, k2T, d2, nb, ns, nois, xh, outX);
    // image skip
    to_image_kernel<<<dim3(W2 / 256, NB), 256, 0, stream>>>(xh, img, imgk, iupk, outIm);
}
